// FrequencyTransformer_26946624815113
// MI455X (gfx1250) — compile-verified, matrix operand order checked by reading, not run on hardware
//
#include <hip/hip_runtime.h>
#include <hip/hip_bf16.h>

// ---------------------------------------------------------------------------
// FrequencyTransformer forward for MI455X (gfx1250), wave32 + WMMA bf16.
//   x:[1,192,192,64] f32 -> out:[1,192,192,64] f32
// Dominant cost: 3x3 conv 768->768 = 391 GFLOP -> implicit GEMM on
// v_wmma_f32_16x16x32_bf16 (f32 accumulate). hidden (56.6MB bf16) + packed
// weights (10.6MB) are L2-resident (192MB). Each conv wave computes a 16x64
// output strip (4 accumulators) so each A fragment feeds 4 WMMAs: ~13 FLOP
// per L2 byte -> matrix-pipe bound, not L2-bound.
// ---------------------------------------------------------------------------

typedef __bf16 bf16;
typedef __attribute__((ext_vector_type(16))) __bf16 v16bf;
typedef __attribute__((ext_vector_type(8)))  __bf16 v8bf;
typedef __attribute__((ext_vector_type(8)))  float  v8f;

#define HH 192
#define WW 192
#define CIN 64
#define HID 768
#define QC 256            // channels of each of q,k,v
#define NPIX (HH*WW)      // 36864
#define MT (NPIX/16)      // 2304 M-tiles
#define KTC 216           // conv K-tiles (9*768/32)

// ---- WMMA fragment helpers (16-bit A/B layout, ISA 7.12.2) ----------------
// per-lane v16bf: elems 0..7 = K in [ (lane>>4)*8 , +8 ), elems 8..15 = +16.
__device__ __forceinline__ v16bf frag_cat(v8bf lo, v8bf hi) {
  return __builtin_shufflevector(lo, hi, 0,1,2,3,4,5,6,7,8,9,10,11,12,13,14,15);
}
__device__ __forceinline__ v16bf load_afrag(const bf16* p) {
  v8bf lo = *(const v8bf*)p;          // K-half at +0
  v8bf hi = *(const v8bf*)(p + 16);   // K-half at +16
  return frag_cat(lo, hi);
}
__device__ __forceinline__ v8f wmma_bf16(v16bf a, v16bf b, v8f c) {
  return __builtin_amdgcn_wmma_f32_16x16x32_bf16(false, a, false, b,
                                                 (short)0, c, false, false);
}

// ---- pack row-major f32 [K,N] weight into B-fragment order (bf16) ----------
// layout: frag f = nt*(K/32)+kt ; element i = (f*32 + lane)*16 + e
__global__ void pack_b_kernel(const float* __restrict__ W, bf16* __restrict__ P,
                              int K, int N) {
  int i = blockIdx.x * blockDim.x + threadIdx.x;
  int Kt = K >> 5;
  int total = Kt * (N >> 4) * 512;
  if (i >= total) return;
  int e    = i & 15;
  int lane = (i >> 4) & 31;
  int f    = i >> 9;
  int kt = f % Kt, nt = f / Kt;
  int khalf = (lane >> 4) * 8;
  int k = kt * 32 + ((e < 8) ? (khalf + e) : (16 + khalf + (e - 8)));
  int n = nt * 16 + (lane & 15);
  P[i] = (bf16)W[(size_t)k * N + n];
}

// ---- LayerNorm over C=64, emit bf16 ---------------------------------------
__global__ void ln1_kernel(const float* __restrict__ X,
                           const float* __restrict__ sc,
                           const float* __restrict__ bi,
                           bf16* __restrict__ XN) {
  int lane = threadIdx.x & 31;
  int row  = blockIdx.x * (blockDim.x >> 5) + (threadIdx.x >> 5);
  const float* xr = X + (size_t)row * CIN;
  float2 v = *(const float2*)(xr + lane * 2);
  float s  = v.x + v.y;
  float s2 = v.x * v.x + v.y * v.y;
  #pragma unroll
  for (int off = 16; off; off >>= 1) {
    s  += __shfl_xor(s,  off, 32);
    s2 += __shfl_xor(s2, off, 32);
  }
  float mean = s * (1.f / 64.f);
  float var  = s2 * (1.f / 64.f) - mean * mean;
  float inv  = rsqrtf(var + 1e-6f);
  int c = lane * 2;
  XN[(size_t)row * CIN + c]     = (bf16)((v.x - mean) * inv * sc[c]     + bi[c]);
  XN[(size_t)row * CIN + c + 1] = (bf16)((v.y - mean) * inv * sc[c + 1] + bi[c + 1]);
}

// ---- GEMM1: hidden[NPIX,768] = xn[NPIX,64] @ w_hidden  (bf16 out) ---------
__global__ __launch_bounds__(256) void gemm_expand_kernel(
    const bf16* __restrict__ A, const bf16* __restrict__ Bp,
    bf16* __restrict__ Hd) {
  int wid  = blockIdx.x * 8 + (threadIdx.x >> 5);
  int lane = threadIdx.x & 31;
  int nt = wid % (HID / 16);          // 48
  int mt = wid / (HID / 16);
  if (mt >= MT) return;
  int m = mt * 16 + (lane & 15);
  v8f acc = {};
  #pragma unroll
  for (int kt = 0; kt < 2; ++kt) {    // K = 64
    v16bf a = load_afrag(A + (size_t)m * CIN + kt * 32 + (lane >> 4) * 8);
    v16bf b = *(const v16bf*)(Bp + (size_t)(nt * 2 + kt) * 512 + lane * 16);
    acc = wmma_bf16(a, b, acc);
  }
  int col   = nt * 16 + (lane & 15);
  int rbase = mt * 16 + (lane >> 4) * 8;
  #pragma unroll
  for (int j = 0; j < 8; ++j)
    Hd[(size_t)(rbase + j) * HID + col] = (bf16)acc[j];
}

// ---- Conv 3x3 SAME 768->768 as implicit GEMM, K = 9*768 = 6912 ------------
// One wave computes a 16(M) x 64(N) strip: 4 accumulators, A reused 4x.
__global__ __launch_bounds__(256) void conv3x3_kernel(
    const bf16* __restrict__ Hd, const bf16* __restrict__ Bp,
    float* __restrict__ QKV) {
  const size_t NSTRIDE = (size_t)KTC * 512;   // packed-B stride per N-tile
  int wid  = blockIdx.x * 8 + (threadIdx.x >> 5);
  int lane = threadIdx.x & 31;
  int ng = wid % 12;                  // group of 4 N-tiles (48/4)
  int mt = wid / 12;
  if (mt >= MT) return;
  int m = mt * 16 + (lane & 15);
  int y = m / WW, x = m % WW;
  int co = (lane >> 4) * 8;
  const bf16* b0 = Bp + lane * 16 + (size_t)(ng * 4) * NSTRIDE;
  v8f acc[4] = {};
  for (int r = 0; r < 9; ++r) {
    int dy = y + r / 3 - 1;
    int dx = x + r % 3 - 1;
    bool inb = ((unsigned)dy < (unsigned)HH) & ((unsigned)dx < (unsigned)WW);
    const bf16* arow = Hd + ((size_t)dy * WW + dx) * HID + co;
    #pragma unroll 2
    for (int cb = 0; cb < 24; ++cb) {     // 24 chunks of K=32 per tap
      int kk = r * 24 + cb;
      __builtin_prefetch((const void*)(b0 + (size_t)(kk + 1) * 512), 0, 1);
      v16bf a = {};
      if (inb) a = load_afrag(arow + cb * 32);   // per-lane zero-fill (SAME pad)
      #pragma unroll
      for (int i = 0; i < 4; ++i) {
        v16bf b = *(const v16bf*)(b0 + i * NSTRIDE + (size_t)kk * 512);
        acc[i] = wmma_bf16(a, b, acc[i]);        // EXEC all-1s here
      }
    }
  }
  int rbase = mt * 16 + (lane >> 4) * 8;
  #pragma unroll
  for (int i = 0; i < 4; ++i) {
    int col = (ng * 4 + i) * 16 + (lane & 15);
    #pragma unroll
    for (int j = 0; j < 8; ++j)
      QKV[(size_t)(rbase + j) * HID + col] = acc[i][j];
  }
}

// ---- per-channel 8x8 circular conv == irfft2(rfft2(q)*rfft2(k)) -----------
// q/k patches staged via async global->LDS copies (ASYNCcnt path).
__global__ void patch_mix_kernel(const float* __restrict__ QKV,
                                 float* __restrict__ MIX) {
  __shared__ float qs[64];
  __shared__ float ks[64];
  int b  = blockIdx.x;                 // (ph*24+pw)*256 + ch
  int ch = b & 255;
  int pb = b >> 8;
  int pw = pb % 24, ph = pb / 24;
  int t  = threadIdx.x;                // 64 threads: (p1,p2)
  int p1 = t >> 3, p2 = t & 7;
  size_t pix = (size_t)(ph * 8 + p1) * WW + (pw * 8 + p2);

  // generic LDS address: low 32 bits are the LDS byte offset
  unsigned qdst = (unsigned)(uintptr_t)&qs[t];
  unsigned kdst = (unsigned)(uintptr_t)&ks[t];
  unsigned long long qsrc = (unsigned long long)(uintptr_t)(QKV + pix * HID + ch);
  unsigned long long ksrc = (unsigned long long)(uintptr_t)(QKV + pix * HID + QC + ch);
  asm volatile("global_load_async_to_lds_b32 %0, %1, off"
               :: "v"(qdst), "v"(qsrc) : "memory");
  asm volatile("global_load_async_to_lds_b32 %0, %1, off"
               :: "v"(kdst), "v"(ksrc) : "memory");
  asm volatile("s_wait_asynccnt 0x0" ::: "memory");
  __syncthreads();

  float acc = 0.f;
  #pragma unroll
  for (int a = 0; a < 8; ++a)
    #pragma unroll
    for (int c = 0; c < 8; ++c)
      acc += qs[a * 8 + c] * ks[((p1 - a) & 7) * 8 + ((p2 - c) & 7)];
  MIX[pix * QC + ch] = acc;
}

// ---- LN over 256 channels, * v, emit bf16 ---------------------------------
__global__ void ln2_mulv_kernel(const float* __restrict__ MIX,
                                const float* __restrict__ QKV,
                                const float* __restrict__ sc,
                                const float* __restrict__ bi,
                                bf16* __restrict__ T) {
  int lane = threadIdx.x & 31;
  int row  = blockIdx.x * (blockDim.x >> 5) + (threadIdx.x >> 5);
  const float* mr = MIX + (size_t)row * QC;
  float mv[8];
  *(float4*)&mv[0] = *(const float4*)(mr + lane * 8);
  *(float4*)&mv[4] = *(const float4*)(mr + lane * 8 + 4);
  float s = 0.f, s2 = 0.f;
  #pragma unroll
  for (int j = 0; j < 8; ++j) { s += mv[j]; s2 += mv[j] * mv[j]; }
  #pragma unroll
  for (int off = 16; off; off >>= 1) {
    s  += __shfl_xor(s,  off, 32);
    s2 += __shfl_xor(s2, off, 32);
  }
  float mean = s * (1.f / 256.f);
  float var  = s2 * (1.f / 256.f) - mean * mean;
  float inv  = rsqrtf(var + 1e-6f);
  const float* vr = QKV + (size_t)row * HID + 2 * QC;   // v channels
  #pragma unroll
  for (int j = 0; j < 8; ++j) {
    int c = lane * 8 + j;
    float ln = (mv[j] - mean) * inv * sc[c] + bi[c];
    T[(size_t)row * QC + c] = (bf16)(ln * vr[c]);
  }
}

// ---- GEMM2: out[NPIX,64] = T[NPIX,256] @ w_out + x (residual) -------------
__global__ __launch_bounds__(256) void gemm_out_kernel(
    const bf16* __restrict__ A, const bf16* __restrict__ Bp,
    const float* __restrict__ X, float* __restrict__ OUT) {
  int wid  = blockIdx.x * 8 + (threadIdx.x >> 5);
  int lane = threadIdx.x & 31;
  int nt = wid % 4;                    // N = 64
  int mt = wid / 4;
  if (mt >= MT) return;
  int m = mt * 16 + (lane & 15);
  v8f acc = {};
  #pragma unroll
  for (int kt = 0; kt < 8; ++kt) {     // K = 256
    v16bf a = load_afrag(A + (size_t)m * QC + kt * 32 + (lane >> 4) * 8);
    v16bf b = *(const v16bf*)(Bp + (size_t)(nt * 8 + kt) * 512 + lane * 16);
    acc = wmma_bf16(a, b, acc);
  }
  int col   = nt * 16 + (lane & 15);
  int rbase = mt * 16 + (lane >> 4) * 8;
  #pragma unroll
  for (int j = 0; j < 8; ++j) {
    size_t idx = (size_t)(rbase + j) * CIN + col;
    OUT[idx] = acc[j] + X[idx];
  }
}

// ---------------------------------------------------------------------------
extern "C" void kernel_launch(void* const* d_in, const int* in_sizes, int n_in,
                              void* d_out, int out_size, void* d_ws, size_t ws_size,
                              hipStream_t stream) {
  (void)in_sizes; (void)n_in; (void)out_size; (void)ws_size;
  const float* x        = (const float*)d_in[0];   // [36864,64]
  const float* w_hidden = (const float*)d_in[1];   // [64,768]
  const float* w_qkv    = (const float*)d_in[2];   // [3,3,768,768] == [6912,768]
  const float* w_out    = (const float*)d_in[3];   // [256,64]
  const float* ln1_s    = (const float*)d_in[4];
  const float* ln1_b    = (const float*)d_in[5];
  const float* ln2_s    = (const float*)d_in[6];
  const float* ln2_b    = (const float*)d_in[7];
  float* out = (float*)d_out;

  // workspace carve-up (256B aligned), ~242 MB total
  char* ws = (char*)d_ws;
  size_t off = 0;
  auto take = [&](size_t bytes) { char* p = ws + off; off += (bytes + 255) & ~(size_t)255; return p; };
  bf16*  xn     = (bf16*)take((size_t)NPIX * CIN * 2);        //  4.7 MB
  bf16*  hidden = (bf16*)take((size_t)NPIX * HID * 2);        // 56.6 MB
  float* qkv    = (float*)take((size_t)NPIX * HID * 4);       // 113.2 MB
  float* mix    = (float*)take((size_t)NPIX * QC * 4);        // 37.7 MB
  bf16*  tbuf   = (bf16*)take((size_t)NPIX * QC * 2);         // 18.9 MB
  bf16*  pwh    = (bf16*)take((size_t)2 * 48 * 512 * 2);      //  0.1 MB
  bf16*  pwqkv  = (bf16*)take((size_t)KTC * 48 * 512 * 2);    // 10.6 MB
  bf16*  pwout  = (bf16*)take((size_t)8 * 4 * 512 * 2);       // 32 KB

  // 1) pack weights into B-fragment layout (bf16)
  pack_b_kernel<<<(2 * 48 * 512 + 255) / 256, 256, 0, stream>>>(w_hidden, pwh, 64, HID);
  pack_b_kernel<<<(KTC * 48 * 512 + 255) / 256, 256, 0, stream>>>(w_qkv, pwqkv, 9 * HID, HID);
  pack_b_kernel<<<(8 * 4 * 512 + 255) / 256, 256, 0, stream>>>(w_out, pwout, QC, CIN);

  // 2) LN1 -> bf16 activations (8 rows / 256-thread block)
  ln1_kernel<<<NPIX / 8, 256, 0, stream>>>(x, ln1_s, ln1_b, xn);

  // 3) expand GEMM 64->768 (one 16x16 tile per wave, 8 waves/block)
  gemm_expand_kernel<<<(MT * 48) / 8, 256, 0, stream>>>(xn, pwh, hidden);

  // 4) 3x3 conv 768->768 implicit GEMM (391 GFLOP; 16x64 strip per wave)
  conv3x3_kernel<<<(MT * 12) / 8, 256, 0, stream>>>(hidden, pwqkv, qkv);

  // 5) spectral patch mixing == per-channel 8x8 circular conv (async->LDS)
  patch_mix_kernel<<<24 * 24 * 256, 64, 0, stream>>>(qkv, mix);

  // 6) LN2 * v -> bf16
  ln2_mulv_kernel<<<NPIX / 8, 256, 0, stream>>>(mix, qkv, ln2_s, ln2_b, tbuf);

  // 7) output GEMM 256->64 fused with residual
  gemm_out_kernel<<<(MT * 4) / 8, 256, 0, stream>>>(tbuf, pwout, x, out);
}
